// MLA_50113678409882
// MI455X (gfx1250) — matrix operand order repaired and checked by reading, not compiled, not run on hardware
//
#include <hip/hip_runtime.h>
#include <hip/hip_bf16.h>

// ---------------------------------------------------------------------------
// DeepSeek-style sparse MLA for MI455X (gfx1250, wave32, WMMA bf16).
// B=2, S=2048, DIM=2048, NH=16, QLR=768, KVLR=512, NOPE=128, ROPE=64,
// VDIM=128, IH=16, IHD=128, ITOPK=512.
// ---------------------------------------------------------------------------

#define C_B     2
#define C_S     2048
#define C_BS    4096            // B*S
#define C_DIM   2048
#define C_NH    16
#define C_QLR   768
#define C_KVLR  512
#define C_NOPE  128
#define C_ROPE  64
#define C_VDIM  128
#define C_IH    16
#define C_IHD   128
#define C_TOPK  512
#define C_QDIM  192             // NOPE+ROPE
#define C_CAT   576             // KVLR+ROPE
#define C_NEG   (-1000000000.0f)

// 1/sqrt(NOPE+ROPE)
#define SCALE_ATT 0.07216878364870322f
// IH^-0.5 * IHD^-0.5 = 1/(4 * sqrt(128))
#define SCALE_IDX 0.022097086912079608f

typedef __bf16 bf16_t;
typedef __attribute__((ext_vector_type(16))) __bf16 bf16x16;
typedef __attribute__((ext_vector_type(8)))  __bf16 bf16x8;
typedef __attribute__((ext_vector_type(8)))  float  f32x8;

union ABfrag { bf16x16 v; bf16x8 h[2]; };

__device__ __forceinline__ bf16_t f2bf(float f) {
    union { float f; unsigned u; } v; v.f = f;
    unsigned r = (v.u + 0x7FFFu + ((v.u >> 16) & 1u)) >> 16;
    union { unsigned short s; bf16_t b; } o; o.s = (unsigned short)r;
    return o.b;
}

__device__ __forceinline__ void storeC(float* p, float v)  { *p = v; }
__device__ __forceinline__ void storeC(bf16_t* p, float v) { *p = f2bf(v); }

#define WMMA_BF16(A, Bm, Cacc) \
    __builtin_amdgcn_wmma_f32_16x16x32_bf16(false, (A), false, (Bm), (short)0, (Cacc), false, false)

// ---------------------------------------------------------------------------
// f32 -> bf16 conversion (grid-stride)
// ---------------------------------------------------------------------------
__global__ void k_cvt(const float* __restrict__ src, bf16_t* __restrict__ dst, size_t n) {
    size_t i = (size_t)blockIdx.x * blockDim.x + threadIdx.x;
    size_t stride = (size_t)gridDim.x * blockDim.x;
    for (; i < n; i += stride) dst[i] = f2bf(src[i]);
}

// ---------------------------------------------------------------------------
// Generic WMMA GEMM: C[m,n] = sum_k A[m*lda+k] * W[n*ldb+k]
// Register-blocked: each wave computes a 32(M) x 64(N) tile = 8 accumulators,
// 8 WMMA per 2 A-fragment + 4 B-fragment loads.
// grid.x = M/32, grid.y = ceil(N/256), block = 128 (4 waves)
// ---------------------------------------------------------------------------
template <typename OutT>
__global__ void k_gemm(const bf16_t* __restrict__ A, int lda,
                       const bf16_t* __restrict__ W, int ldb,
                       OutT* __restrict__ C, int ldc, int N, int K) {
    int m0 = blockIdx.x * 32;
    int wv = threadIdx.x >> 5;
    int n0 = blockIdx.y * 256 + wv * 64;
    if (n0 >= N) return;
    int lane = threadIdx.x & 31;
    int r = lane & 15, hi = lane >> 4;

    const bf16_t* pa0 = A + (size_t)(m0 + r) * lda + hi * 8;
    const bf16_t* pa1 = A + (size_t)(m0 + 16 + r) * lda + hi * 8;
    const bf16_t* pb[4];
    bool valid[4];
#pragma unroll
    for (int ni = 0; ni < 4; ++ni) {
        int n_i = n0 + ni * 16;
        valid[ni] = (n_i < N);
        int ncl = valid[ni] ? n_i : 0;
        pb[ni] = W + (size_t)(ncl + r) * ldb + hi * 8;
    }

    f32x8 acc[2][4];
#pragma unroll
    for (int mi = 0; mi < 2; ++mi)
#pragma unroll
        for (int ni = 0; ni < 4; ++ni)
#pragma unroll
            for (int j = 0; j < 8; ++j) acc[mi][ni][j] = 0.f;

    for (int k0 = 0; k0 < K; k0 += 32) {
        if (k0 + 64 < K) {
            __builtin_prefetch(pa0 + k0 + 64, 0, 1);   // -> global_prefetch_b8
            __builtin_prefetch(pa1 + k0 + 64, 0, 1);
        }
        ABfrag a0, a1;
        a0.h[0] = *(const bf16x8*)(pa0 + k0);
        a0.h[1] = *(const bf16x8*)(pa0 + k0 + 16);
        a1.h[0] = *(const bf16x8*)(pa1 + k0);
        a1.h[1] = *(const bf16x8*)(pa1 + k0 + 16);
#pragma unroll
        for (int ni = 0; ni < 4; ++ni) {
            ABfrag b;
            b.h[0] = *(const bf16x8*)(pb[ni] + k0);
            b.h[1] = *(const bf16x8*)(pb[ni] + k0 + 16);
            acc[0][ni] = WMMA_BF16(a0.v, b.v, acc[0][ni]);
            acc[1][ni] = WMMA_BF16(a1.v, b.v, acc[1][ni]);
        }
    }
#pragma unroll
    for (int ni = 0; ni < 4; ++ni) {
        if (!valid[ni]) continue;
        int n = n0 + ni * 16 + r;
#pragma unroll
        for (int mi = 0; mi < 2; ++mi)
#pragma unroll
            for (int i = 0; i < 8; ++i)
                storeC(&C[(size_t)(m0 + mi * 16 + hi * 8 + i) * ldc + n], acc[mi][ni][i]);
    }
}

// ---------------------------------------------------------------------------
// RMSNorm over 768 (qr): f32 in -> bf16 out.  block=256, 3 elems/thread
// ---------------------------------------------------------------------------
__global__ void k_rmsnorm_qr(const float* __restrict__ in, const float* __restrict__ w,
                             bf16_t* __restrict__ out) {
    const float* row = in + (size_t)blockIdx.x * C_QLR;
    __shared__ float red[256];
    int tid = threadIdx.x;
    float v0 = row[tid], v1 = row[tid + 256], v2 = row[tid + 512];
    red[tid] = v0 * v0 + v1 * v1 + v2 * v2;
    __syncthreads();
    for (int s = 128; s > 0; s >>= 1) { if (tid < s) red[tid] += red[tid + s]; __syncthreads(); }
    float r = rsqrtf(red[0] * (1.0f / C_QLR) + 1e-6f);
    bf16_t* o = out + (size_t)blockIdx.x * C_QLR;
    o[tid]       = f2bf(v0 * r * w[tid]);
    o[tid + 256] = f2bf(v1 * r * w[tid + 256]);
    o[tid + 512] = f2bf(v2 * r * w[tid + 512]);
}

// ---------------------------------------------------------------------------
// kv pack: rmsnorm(kv[0:512]) and rope(k_pe[512:576]) -> kcat bf16 (BS,576)
// ---------------------------------------------------------------------------
__global__ void k_kv_pack(const float* __restrict__ kvfull, const float* __restrict__ w,
                          const float* __restrict__ cosb, const float* __restrict__ sinb,
                          bf16_t* __restrict__ kcat) {
    int bs = blockIdx.x;
    const float* row = kvfull + (size_t)bs * C_CAT;
    __shared__ float red[256];
    int tid = threadIdx.x;
    float v0 = row[tid], v1 = row[tid + 256];
    red[tid] = v0 * v0 + v1 * v1;
    __syncthreads();
    for (int s = 128; s > 0; s >>= 1) { if (tid < s) red[tid] += red[tid + s]; __syncthreads(); }
    float r = rsqrtf(red[0] * (1.0f / C_KVLR) + 1e-6f);
    bf16_t* o = kcat + (size_t)bs * C_CAT;
    o[tid]       = f2bf(v0 * r * w[tid]);
    o[tid + 256] = f2bf(v1 * r * w[tid + 256]);
    if (tid < 32) {
        int s = bs & (C_S - 1);
        float x1 = row[512 + tid], x2 = row[544 + tid];
        float c = cosb[s * 32 + tid], sn = sinb[s * 32 + tid];
        o[512 + tid] = f2bf(x1 * c - x2 * sn);
        o[544 + tid] = f2bf(x1 * sn + x2 * c);
    }
}

// ---------------------------------------------------------------------------
// q pack: split q_f (BS, 16*192) -> qnope bf16 (BS,16,128) and roped q_pe
// into qcat columns [512:576)   (qcat is (BS*16, 576))
// ---------------------------------------------------------------------------
__global__ void k_q_pack(const float* __restrict__ qf_all, const float* __restrict__ cosb,
                         const float* __restrict__ sinb, bf16_t* __restrict__ qnope,
                         bf16_t* __restrict__ qcat) {
    int bs = blockIdx.x;
    int s = bs & (C_S - 1);
    const float* qf = qf_all + (size_t)bs * (C_NH * C_QDIM);
    int tid = threadIdx.x;
    bf16_t* qn = qnope + (size_t)bs * (C_NH * C_NOPE);
    for (int i = tid; i < C_NH * C_NOPE; i += 256) {
        int h = i >> 7, d = i & 127;
        qn[i] = f2bf(qf[h * C_QDIM + d]);
    }
    bf16_t* qc = qcat + (size_t)bs * C_NH * C_CAT;
    for (int i = tid; i < C_NH * 32; i += 256) {
        int h = i >> 5, j = i & 31;
        float x1 = qf[h * C_QDIM + 128 + j], x2 = qf[h * C_QDIM + 160 + j];
        float c = cosb[s * 32 + j], sn = sinb[s * 32 + j];
        qc[(size_t)h * C_CAT + 512 + j] = f2bf(x1 * c - x2 * sn);
        qc[(size_t)h * C_CAT + 544 + j] = f2bf(x1 * sn + x2 * c);
    }
}

// ---------------------------------------------------------------------------
// indexer q pack: rope first 64 dims of each 128-dim head, copy rest. bf16 out.
// ---------------------------------------------------------------------------
__global__ void k_qidx_pack(const float* __restrict__ qf_all, const float* __restrict__ cosb,
                            const float* __restrict__ sinb, bf16_t* __restrict__ out_all) {
    int bs = blockIdx.x;
    int s = bs & (C_S - 1);
    const float* qf = qf_all + (size_t)bs * (C_IH * C_IHD);
    bf16_t* out = out_all + (size_t)bs * (C_IH * C_IHD);
    int tid = threadIdx.x;
    for (int i = tid; i < C_IH * 32; i += 256) {
        int h = i >> 5, j = i & 31;
        float x1 = qf[h * C_IHD + j], x2 = qf[h * C_IHD + 32 + j];
        float c = cosb[s * 32 + j], sn = sinb[s * 32 + j];
        out[h * C_IHD + j]      = f2bf(x1 * c - x2 * sn);
        out[h * C_IHD + 32 + j] = f2bf(x1 * sn + x2 * c);
    }
    for (int i = tid; i < C_IH * 64; i += 256) {
        int h = i >> 6, d = 64 + (i & 63);
        out[h * C_IHD + d] = f2bf(qf[h * C_IHD + d]);
    }
}

// ---------------------------------------------------------------------------
// indexer k pack: layernorm(128) + rope first 64. block=128.
// ---------------------------------------------------------------------------
__global__ void k_kidx_pack(const float* __restrict__ kf, const float* __restrict__ gw,
                            const float* __restrict__ gb, const float* __restrict__ cosb,
                            const float* __restrict__ sinb, bf16_t* __restrict__ out_all) {
    int bs = blockIdx.x;
    int s = bs & (C_S - 1);
    int tid = threadIdx.x;
    __shared__ float red[128];
    __shared__ float sv[128];
    float v = kf[(size_t)bs * C_IHD + tid];
    red[tid] = v; __syncthreads();
    for (int st = 64; st > 0; st >>= 1) { if (tid < st) red[tid] += red[tid + st]; __syncthreads(); }
    float m = red[0] * (1.0f / C_IHD);
    __syncthreads();
    float d = v - m;
    red[tid] = d * d; __syncthreads();
    for (int st = 64; st > 0; st >>= 1) { if (tid < st) red[tid] += red[tid + st]; __syncthreads(); }
    float var = red[0] * (1.0f / C_IHD);
    sv[tid] = d * rsqrtf(var + 1e-6f) * gw[tid] + gb[tid];
    __syncthreads();
    bf16_t* out = out_all + (size_t)bs * C_IHD;
    if (tid < 32) {
        float x1 = sv[tid], x2 = sv[tid + 32];
        float c = cosb[s * 32 + tid], sn = sinb[s * 32 + tid];
        out[tid]      = f2bf(x1 * c - x2 * sn);
        out[tid + 32] = f2bf(x1 * sn + x2 * c);
    } else if (tid >= 64) {
        out[tid] = f2bf(sv[tid]);
    }
}

// ---------------------------------------------------------------------------
// wkv_b preparation:
//  w1t[h][c][d] = wkv_b[h*256 + d][c]        (d<128)  -> for q_abs GEMM (N=512,K=128)
//  w2 [h][d][c] = wkv_b[h*256 + 128 + d][c]  (d<128)  -> for out GEMM   (N=128,K=512)
// ---------------------------------------------------------------------------
__global__ void k_prep_wkv_b(const float* __restrict__ wkv, bf16_t* __restrict__ w1t,
                             bf16_t* __restrict__ w2) {
    int r = blockIdx.x;             // 0..4095 rows of wkv_b
    int h = r >> 8, rr = r & 255;
    int tid = threadIdx.x;
    const float* src = wkv + (size_t)r * C_KVLR;
    if (rr < 128) {
        int d = rr;
        bf16_t* dst = w1t + (size_t)h * C_KVLR * C_NOPE;
        for (int c = tid; c < C_KVLR; c += 256) dst[(size_t)c * C_NOPE + d] = f2bf(src[c]);
    } else {
        int d = rr - 128;
        bf16_t* dst = w2 + (size_t)h * C_VDIM * C_KVLR + (size_t)d * C_KVLR;
        for (int c = tid; c < C_KVLR; c += 256) dst[c] = f2bf(src[c]);
    }
}

// ---------------------------------------------------------------------------
// Indexer score: iscore[b,s,t] = sum_h relu(q_idx[b,s,h,:]·k_idx[b,t,:]) * w[b,s,h] + mask[s,t]
// grid = (S/16, S/128, B), block = 256 (8 waves, wave -> one 16x16 (s,t) tile)
// B-fragments (k_idx) are loop-invariant across heads -> stay in VGPRs.
// ---------------------------------------------------------------------------
__global__ void k_idx_score(const bf16_t* __restrict__ qidx, const bf16_t* __restrict__ kidx,
                            const float* __restrict__ wts, const float* __restrict__ mask,
                            float* __restrict__ iscore) {
    int b = blockIdx.z;
    int s0 = blockIdx.x * 16;
    int tg = blockIdx.y;
    __shared__ float wsh[256];      // [s-row(16)][head(16)]
    int tid = threadIdx.x;
    wsh[tid] = wts[((size_t)(b * C_S) + s0 + (tid >> 4)) * C_IH + (tid & 15)] * SCALE_IDX;
    __syncthreads();

    int lane = tid & 31, wv = tid >> 5;
    int t0 = tg * 128 + wv * 16;
    int r = lane & 15, hi = lane >> 4;

    const bf16_t* qbase = qidx + ((size_t)(b * C_S + s0 + r) * C_IH) * C_IHD;
    const bf16_t* kbase = kidx + (size_t)(b * C_S + t0 + r) * C_IHD;

    float out[8] = {0.f,0.f,0.f,0.f,0.f,0.f,0.f,0.f};
    for (int h = 0; h < C_IH; ++h) {
        f32x8 acc = {0.f,0.f,0.f,0.f,0.f,0.f,0.f,0.f};
        const bf16_t* qh = qbase + (size_t)h * C_IHD;
#pragma unroll
        for (int k0 = 0; k0 < C_IHD; k0 += 32) {
            ABfrag a, bb;
            a.h[0] = *(const bf16x8*)(qh + k0 + hi * 16);
            a.h[1] = *(const bf16x8*)(qh + k0 + hi * 16 + 8);
            bb.v   = *(const bf16x16*)(kbase + k0 + hi * 16);
            acc = WMMA_BF16(a.v, bb.v, acc);
        }
#pragma unroll
        for (int i = 0; i < 8; ++i)
            out[i] += fmaxf(acc[i], 0.f) * wsh[(hi * 8 + i) * 16 + h];
    }
    for (int i = 0; i < 8; ++i) {
        int s = s0 + hi * 8 + i, t = t0 + r;
        iscore[((size_t)b * C_S + s) * C_S + t] = out[i] + mask[(size_t)s * C_S + t];
    }
}

// ---------------------------------------------------------------------------
// Top-512 per row via deterministic 4-pass radix select (MSB-first, 8b digits)
// grid = BS, block = 256
// ---------------------------------------------------------------------------
__device__ __forceinline__ unsigned fkey(float f) {
    unsigned u = __float_as_uint(f);
    return (u & 0x80000000u) ? ~u : (u | 0x80000000u);    // ascending order key
}

__global__ void k_topk(const float* __restrict__ iscore, int* __restrict__ topk) {
    int bs = blockIdx.x;
    const float* row = iscore + (size_t)bs * C_S;
    __shared__ int hist[256];
    __shared__ unsigned s_prefix;
    __shared__ int s_remain;
    __shared__ int cg[256], ce[256];
    int tid = threadIdx.x;
    if (tid == 0) { s_prefix = 0u; s_remain = C_TOPK; }
    __syncthreads();

    for (int pass = 0; pass < 4; ++pass) {
        int shift = 24 - pass * 8;
        hist[tid] = 0;
        __syncthreads();
        unsigned prefix = s_prefix;
        unsigned pmask = (pass == 0) ? 0u : (0xFFFFFFFFu << (shift + 8));
        for (int i = tid; i < C_S; i += 256) {
            unsigned k = fkey(row[i]);
            if ((k & pmask) == prefix) atomicAdd(&hist[(k >> shift) & 255], 1);
        }
        __syncthreads();
        if (tid == 0) {
            int remain = s_remain;
            int c = 0, bkt = 255;
            for (; bkt >= 0; --bkt) { c += hist[bkt]; if (c >= remain) break; }
            s_remain = remain - (c - hist[bkt]);          // still needed within bucket
            s_prefix = prefix | ((unsigned)bkt << shift);
        }
        __syncthreads();
    }
    unsigned T = s_prefix;
    int need_eq = s_remain;
    int ngt = C_TOPK - need_eq;

    int base = tid * 8;
    unsigned keys[8];
    int myg = 0, mye = 0;
    for (int j = 0; j < 8; ++j) {
        keys[j] = fkey(row[base + j]);
        if (keys[j] > T) myg++; else if (keys[j] == T) mye++;
    }
    cg[tid] = myg; ce[tid] = mye;
    __syncthreads();
    if (tid == 0) {
        int ag = 0, ae = 0;
        for (int i = 0; i < 256; ++i) {
            int g = cg[i], e = ce[i];
            cg[i] = ag; ce[i] = ae;
            ag += g; ae += e;
        }
    }
    __syncthreads();
    int pg = cg[tid], pe = ce[tid];
    int* out = topk + (size_t)bs * C_TOPK;
    for (int j = 0; j < 8; ++j) {
        if (keys[j] > T) out[pg++] = base + j;
        else if (keys[j] == T) { if (pe < need_eq) out[ngt + pe] = base + j; pe++; }
    }
}

// ---------------------------------------------------------------------------
// Sparse attention phase 1: scores over the 512 selected keys.
// probs[bs,h,slot] = (Qcat[bs,h,:] · Kcat[b,sel[slot],:]) * scale + mask[s,sel]
// grid = (2, BS), block = 128: wave -> 16h x 64slot (4 tiles share one A-frag)
// ---------------------------------------------------------------------------
__global__ void k_sattn_scores(const bf16_t* __restrict__ qcat, const bf16_t* __restrict__ kcat,
                               const int* __restrict__ topk, const float* __restrict__ mask,
                               float* __restrict__ probs) {
    int bs = blockIdx.y;
    int b = bs >> 11, s = bs & (C_S - 1);
    const int* sel = topk + (size_t)bs * C_TOPK;
    int tid = threadIdx.x, lane = tid & 31, wv = tid >> 5;
    int slot0 = blockIdx.x * 256 + wv * 64;
    int r = lane & 15, hi = lane >> 4;

    const bf16_t* qbase = qcat + ((size_t)bs * C_NH + r) * C_CAT;
    int tsel[4];
    const bf16_t* kb[4];
#pragma unroll
    for (int ni = 0; ni < 4; ++ni) {
        tsel[ni] = sel[slot0 + ni * 16 + r];
        kb[ni] = kcat + ((size_t)(b * C_S) + tsel[ni]) * C_CAT;
    }

    f32x8 acc[4];
#pragma unroll
    for (int ni = 0; ni < 4; ++ni)
#pragma unroll
        for (int j = 0; j < 8; ++j) acc[ni][j] = 0.f;

    for (int k0 = 0; k0 < C_CAT; k0 += 32) {
        ABfrag a;
        a.h[0] = *(const bf16x8*)(qbase + k0 + hi * 16);
        a.h[1] = *(const bf16x8*)(qbase + k0 + hi * 16 + 8);
#pragma unroll
        for (int ni = 0; ni < 4; ++ni) {
            ABfrag bb;
            bb.v = *(const bf16x16*)(kb[ni] + k0 + hi * 16);
            acc[ni] = WMMA_BF16(a.v, bb.v, acc[ni]);
        }
    }
    float* out = probs + (size_t)bs * C_NH * C_TOPK;
#pragma unroll
    for (int ni = 0; ni < 4; ++ni) {
        float msk = mask[(size_t)s * C_S + tsel[ni]];
        for (int i = 0; i < 8; ++i) {
            int h = hi * 8 + i;
            out[(size_t)h * C_TOPK + slot0 + ni * 16 + r] = acc[ni][i] * SCALE_ATT + msk;
        }
    }
}

// ---------------------------------------------------------------------------
// Softmax over 512 slots per (bs,h).  grid = BS*16, block = 256
// ---------------------------------------------------------------------------
__global__ void k_softmax(float* __restrict__ probs) {
    float* row = probs + (size_t)blockIdx.x * C_TOPK;
    __shared__ float red[256];
    int tid = threadIdx.x;
    float a = row[tid], bb = row[tid + 256];
    red[tid] = fmaxf(a, bb);
    __syncthreads();
    for (int st = 128; st > 0; st >>= 1) { if (tid < st) red[tid] = fmaxf(red[tid], red[tid + st]); __syncthreads(); }
    float m = red[0];
    __syncthreads();
    float e0 = __expf(a - m), e1 = __expf(bb - m);
    red[tid] = e0 + e1;
    __syncthreads();
    for (int st = 128; st > 0; st >>= 1) { if (tid < st) red[tid] += red[tid + st]; __syncthreads(); }
    float inv = 1.0f / red[0];
    row[tid] = e0 * inv;
    row[tid + 256] = e1 * inv;
}

// ---------------------------------------------------------------------------
// Sparse attention phase 3: omid[bs,h,c] = sum_slot p[h,slot] * kv[b,sel[slot],c]
// grid = BS, block = 256 (8 waves).  Gathered KV chunk (32t x 512c) staged
// transposed in LDS so B-fragments are contiguous LDS reads.
// ---------------------------------------------------------------------------
__global__ void k_sattn_out(const float* __restrict__ probs, const bf16_t* __restrict__ kcat,
                            const int* __restrict__ topk, bf16_t* __restrict__ omid) {
    int bs = blockIdx.x;
    int b = bs >> 11;
    const int* sel = topk + (size_t)bs * C_TOPK;
    __shared__ bf16_t kvT[C_KVLR * 32];    // [c][t-in-chunk]  32 KB
    __shared__ bf16_t pA[C_NH * 32];       // [h][t-in-chunk]   1 KB
    int tid = threadIdx.x, lane = tid & 31, wv = tid >> 5;
    int r = lane & 15, hi = lane >> 4;
    int c0 = wv * 64;

    f32x8 acc0 = {0.f,0.f,0.f,0.f,0.f,0.f,0.f,0.f};
    f32x8 acc1 = acc0, acc2 = acc0, acc3 = acc0;

    for (int ch = 0; ch < 16; ++ch) {
        // stage gathered KV chunk, transposed
        {
            int tt = tid & 31, cb = (tid >> 5) * 64;
            const bf16_t* src = kcat + ((size_t)(b * C_S) + sel[ch * 32 + tt]) * C_CAT + cb;
            for (int j = 0; j < 64; ++j) kvT[(cb + j) * 32 + tt] = src[j];
        }
        // stage probability chunk as bf16 [h][t]
        {
            int i0 = tid, i1 = tid + 256;
            pA[i0] = f2bf(probs[((size_t)bs * C_NH + (i0 >> 5)) * C_TOPK + ch * 32 + (i0 & 31)]);
            pA[i1] = f2bf(probs[((size_t)bs * C_NH + (i1 >> 5)) * C_TOPK + ch * 32 + (i1 & 31)]);
        }
        __syncthreads();

        ABfrag a;
        a.h[0] = *(const bf16x8*)(pA + r * 32 + hi * 16);
        a.h[1] = *(const bf16x8*)(pA + r * 32 + hi * 16 + 8);
        ABfrag b0, b1, b2, b3;
        b0.v = *(const bf16x16*)(kvT + (c0 + r) * 32 + hi * 16);
        b1.v = *(const bf16x16*)(kvT + (c0 + 16 + r) * 32 + hi * 16);
        b2.v = *(const bf16x16*)(kvT + (c0 + 32 + r) * 32 + hi * 16);
        b3.v = *(const bf16x16*)(kvT + (c0 + 48 + r) * 32 + hi * 16);
        acc0 = WMMA_BF16(a.v, b0.v, acc0);
        acc1 = WMMA_BF16(a.v, b1.v, acc1);
        acc2 = WMMA_BF16(a.v, b2.v, acc2);
        acc3 = WMMA_BF16(a.v, b3.v, acc3);
        __syncthreads();
    }

    bf16_t* out = omid + (size_t)bs * C_NH * C_KVLR;
    for (int i = 0; i < 8; ++i) {
        int h = hi * 8 + i;
        out[(size_t)h * C_KVLR + c0 + r]      = f2bf(acc0[i]);
        out[(size_t)h * C_KVLR + c0 + 16 + r] = f2bf(acc1[i]);
        out[(size_t)h * C_KVLR + c0 + 32 + r] = f2bf(acc2[i]);
        out[(size_t)h * C_KVLR + c0 + 48 + r] = f2bf(acc3[i]);
    }
}

// ---------------------------------------------------------------------------
// Host driver
// ---------------------------------------------------------------------------
static inline int cdiv(int a, int b) { return (a + b - 1) / b; }

extern "C" void kernel_launch(void* const* d_in, const int* in_sizes, int n_in,
                              void* d_out, int out_size, void* d_ws, size_t ws_size,
                              hipStream_t stream) {
    const float* x       = (const float*)d_in[0];
    const float* cosb    = (const float*)d_in[1];
    const float* sinb    = (const float*)d_in[2];
    const float* mask    = (const float*)d_in[3];
    const float* wq_a    = (const float*)d_in[4];
    const float* q_norm  = (const float*)d_in[5];
    const float* wq_b    = (const float*)d_in[6];
    const float* wkv_a   = (const float*)d_in[7];
    const float* kv_norm = (const float*)d_in[8];
    const float* wkv_b   = (const float*)d_in[9];
    const float* wo      = (const float*)d_in[10];
    const float* iwqb    = (const float*)d_in[11];
    const float* iwk     = (const float*)d_in[12];
    const float* iknw    = (const float*)d_in[13];
    const float* iknb    = (const float*)d_in[14];
    const float* iwp     = (const float*)d_in[15];
    float* out = (float*)d_out;

    char* wsp = (char*)d_ws;
    auto alloc = [&](size_t bytes) { char* p = wsp; wsp += (bytes + 255) & ~(size_t)255; return p; };

    bf16_t* xb      = (bf16_t*)alloc((size_t)C_BS * C_DIM * 2);
    bf16_t* wab     = (bf16_t*)alloc((size_t)C_QLR * C_DIM * 2);
    bf16_t* wbb     = (bf16_t*)alloc((size_t)C_NH * C_QDIM * C_QLR * 2);
    bf16_t* wkvab   = (bf16_t*)alloc((size_t)C_CAT * C_DIM * 2);
    bf16_t* iwqbb   = (bf16_t*)alloc((size_t)C_IH * C_IHD * C_QLR * 2);
    bf16_t* iwkb    = (bf16_t*)alloc((size_t)C_IHD * C_DIM * 2);
    bf16_t* iwpb    = (bf16_t*)alloc((size_t)C_IH * C_DIM * 2);
    bf16_t* wob     = (bf16_t*)alloc((size_t)C_DIM * C_DIM * 2);
    bf16_t* w1t     = (bf16_t*)alloc((size_t)C_NH * C_KVLR * C_NOPE * 2);
    bf16_t* w2      = (bf16_t*)alloc((size_t)C_NH * C_VDIM * C_KVLR * 2);
    float*  qr_f    = (float*)alloc((size_t)C_BS * C_QLR * 4);
    bf16_t* qr_bf   = (bf16_t*)alloc((size_t)C_BS * C_QLR * 2);
    float*  q_f     = (float*)alloc((size_t)C_BS * C_NH * C_QDIM * 4);
    float*  kvfull  = (float*)alloc((size_t)C_BS * C_CAT * 4);
    bf16_t* kcat    = (bf16_t*)alloc((size_t)C_BS * C_CAT * 2);
    bf16_t* qnope   = (bf16_t*)alloc((size_t)C_BS * C_NH * C_NOPE * 2);
    bf16_t* qcat    = (bf16_t*)alloc((size_t)C_BS * C_NH * C_CAT * 2);
    float*  qidx_f  = (float*)alloc((size_t)C_BS * C_IH * C_IHD * 4);
    bf16_t* qidx_bf = (bf16_t*)alloc((size_t)C_BS * C_IH * C_IHD * 2);
    float*  kidx_f  = (float*)alloc((size_t)C_BS * C_IHD * 4);
    bf16_t* kidx_bf = (bf16_t*)alloc((size_t)C_BS * C_IHD * 2);
    float*  wts_f   = (float*)alloc((size_t)C_BS * C_IH * 4);
    float*  iscore  = (float*)alloc((size_t)C_B * C_S * C_S * 4);
    int*    topki   = (int*)alloc((size_t)C_BS * C_TOPK * 4);
    float*  probs   = (float*)alloc((size_t)C_BS * C_NH * C_TOPK * 4);
    bf16_t* omid    = (bf16_t*)alloc((size_t)C_BS * C_NH * C_KVLR * 2);
    bf16_t* oproj   = (bf16_t*)alloc((size_t)C_BS * C_NH * C_VDIM * 2);

    // ---- weight / input conversions to bf16 ----
    k_cvt<<<2048, 256, 0, stream>>>(x, xb, (size_t)C_BS * C_DIM);
    k_cvt<<<1024, 256, 0, stream>>>(wq_a, wab, (size_t)C_QLR * C_DIM);
    k_cvt<<<1024, 256, 0, stream>>>(wq_b, wbb, (size_t)C_NH * C_QDIM * C_QLR);
    k_cvt<<<1024, 256, 0, stream>>>(wkv_a, wkvab, (size_t)C_CAT * C_DIM);
    k_cvt<<<1024, 256, 0, stream>>>(iwqb, iwqbb, (size_t)C_IH * C_IHD * C_QLR);
    k_cvt<<<512, 256, 0, stream>>>(iwk, iwkb, (size_t)C_IHD * C_DIM);
    k_cvt<<<128, 256, 0, stream>>>(iwp, iwpb, (size_t)C_IH * C_DIM);
    k_cvt<<<2048, 256, 0, stream>>>(wo, wob, (size_t)C_DIM * C_DIM);
    k_prep_wkv_b<<<C_NH * 256, 256, 0, stream>>>(wkv_b, w1t, w2);

    // ---- projections ----
    k_gemm<float><<<dim3(C_BS / 32, cdiv(C_QLR, 256)), 128, 0, stream>>>(
        xb, C_DIM, wab, C_DIM, qr_f, C_QLR, C_QLR, C_DIM);
    k_rmsnorm_qr<<<C_BS, 256, 0, stream>>>(qr_f, q_norm, qr_bf);
    k_gemm<float><<<dim3(C_BS / 32, cdiv(C_NH * C_QDIM, 256)), 128, 0, stream>>>(
        qr_bf, C_QLR, wbb, C_QLR, q_f, C_NH * C_QDIM, C_NH * C_QDIM, C_QLR);
    k_gemm<float><<<dim3(C_BS / 32, cdiv(C_CAT, 256)), 128, 0, stream>>>(
        xb, C_DIM, wkvab, C_DIM, kvfull, C_CAT, C_CAT, C_DIM);
    k_kv_pack<<<C_BS, 256, 0, stream>>>(kvfull, kv_norm, cosb, sinb, kcat);
    k_q_pack<<<C_BS, 256, 0, stream>>>(q_f, cosb, sinb, qnope, qcat);

    // q_abs per head: qcat[:, h, 0:512] = qnope_h (4096x128) @ w1t_h^T (512x128)
    for (int h = 0; h < C_NH; ++h) {
        k_gemm<bf16_t><<<dim3(C_BS / 32, cdiv(C_KVLR, 256)), 128, 0, stream>>>(
            qnope + h * C_NOPE, C_NH * C_NOPE,
            w1t + (size_t)h * C_KVLR * C_NOPE, C_NOPE,
            qcat + (size_t)h * C_CAT, C_NH * C_CAT, C_KVLR, C_NOPE);
    }

    // ---- indexer ----
    k_gemm<float><<<dim3(C_BS / 32, cdiv(C_IH * C_IHD, 256)), 128, 0, stream>>>(
        qr_bf, C_QLR, iwqbb, C_QLR, qidx_f, C_IH * C_IHD, C_IH * C_IHD, C_QLR);
    k_qidx_pack<<<C_BS, 256, 0, stream>>>(qidx_f, cosb, sinb, qidx_bf);
    k_gemm<float><<<dim3(C_BS / 32, cdiv(C_IHD, 256)), 128, 0, stream>>>(
        xb, C_DIM, iwkb, C_DIM, kidx_f, C_IHD, C_IHD, C_DIM);
    k_kidx_pack<<<C_BS, 128, 0, stream>>>(kidx_f, iknw, iknb, cosb, sinb, kidx_bf);
    k_gemm<float><<<dim3(C_BS / 32, 1), 128, 0, stream>>>(
        xb, C_DIM, iwpb, C_DIM, wts_f, C_IH, C_IH, C_DIM);
    k_idx_score<<<dim3(C_S / 16, C_S / 128, C_B), 256, 0, stream>>>(
        qidx_bf, kidx_bf, wts_f, mask, iscore);
    k_topk<<<C_BS, 256, 0, stream>>>(iscore, topki);

    // ---- sparse attention over top-512 keys ----
    k_sattn_scores<<<dim3(2, C_BS), 128, 0, stream>>>(qcat, kcat, topki, mask, probs);
    k_softmax<<<C_BS * C_NH, 256, 0, stream>>>(probs);
    k_sattn_out<<<C_BS, 256, 0, stream>>>(probs, kcat, topki, omid);

    // per-head output projection: oproj[:, h*128:(h+1)*128] = omid_h @ w2_h^T
    for (int h = 0; h < C_NH; ++h) {
        k_gemm<bf16_t><<<dim3(C_BS / 32, cdiv(C_VDIM, 256)), 128, 0, stream>>>(
            omid + (size_t)h * C_KVLR, C_NH * C_KVLR,
            w2 + (size_t)h * C_VDIM * C_KVLR, C_KVLR,
            oproj + (size_t)h * C_VDIM, C_NH * C_VDIM, C_VDIM, C_KVLR);
    }

    // final: out = oproj (4096x2048) @ wo^T (2048x2048)
    k_gemm<float><<<dim3(C_BS / 32, cdiv(C_DIM, 256)), 128, 0, stream>>>(
        oproj, C_DIM, wob, C_DIM, out, C_DIM, C_DIM, C_DIM);

    (void)in_sizes; (void)n_in; (void)out_size; (void)ws_size;
}
